// MultiHeadSelfAttention_3169685865394
// MI455X (gfx1250) — matrix operand order repaired and checked by reading, not compiled, not run on hardware
//
#include <hip/hip_runtime.h>

// ---------------- problem constants ----------------
#define BB 4
#define CC 2048
#define FF 1024
#define HH 8
#define DH 128   // FF/HH

typedef __attribute__((ext_vector_type(16))) __bf16          v16bf;
typedef __attribute__((ext_vector_type(16))) unsigned short  v16s;
typedef __attribute__((ext_vector_type(8)))  unsigned short  v8s;
typedef __attribute__((ext_vector_type(8)))  float           v8f;
typedef __attribute__((ext_vector_type(4)))  float           v4f;
typedef __attribute__((ext_vector_type(4)))  int             v4i;

#if defined(__has_builtin)
#if __has_builtin(__builtin_amdgcn_global_load_async_to_lds_b128)
#define HAVE_ASYNC_LDS 1
#endif
#endif

// float -> bf16 with round-to-nearest-even
__device__ __forceinline__ unsigned short f2bf(float f) {
  unsigned int u = __builtin_bit_cast(unsigned int, f);
  unsigned int r = u + 0x7FFFu + ((u >> 16) & 1u);
  return (unsigned short)(r >> 16);
}

__device__ __forceinline__ v16bf frag_from(v8s lo, v8s hi) {
  v16s s;
#pragma unroll
  for (int i = 0; i < 8; ++i) { s[i] = lo[i]; s[i + 8] = hi[i]; }
  return __builtin_bit_cast(v16bf, s);
}

// A fragment (16x32 bf16), source row-major MxK (K contiguous), ld in elements.
// Lane m (0..15): row m, K = k0 + {0..7, 16..23}; lane m+16: K = k0 + {8..15, 24..31}.
__device__ __forceinline__ v16bf load_a_bf16(const unsigned short* base, int ld,
                                             int m0, int k0, int lane) {
  const unsigned short* p =
      base + (size_t)(m0 + (lane & 15)) * ld + k0 + ((lane >> 4) << 3);
  v8s lo = *(const v8s*)(p);
  v8s hi = *(const v8s*)(p + 16);
  return frag_from(lo, hi);
}

// B fragment (32x16 bf16), source stored N-major: element (k,n) at base[n*ld + k].
// Lane n (0..15): col n, K = k0 + {0..7, 16..23}; lane n+16: col n, K = k0 + {8..15, 24..31}.
__device__ __forceinline__ v16bf load_b_bf16(const unsigned short* base, int ld,
                                             int n0, int k0, int lane) {
  const unsigned short* p =
      base + (size_t)(n0 + (lane & 15)) * ld + k0 + ((lane >> 4) << 3);
  v8s lo = *(const v8s*)(p);
  v8s hi = *(const v8s*)(p + 16);
  return frag_from(lo, hi);
}

// A fragment built from fp32 source (row-major MxK), converted to bf16 on the fly.
__device__ __forceinline__ v16bf load_a_f32(const float* base, int ld,
                                            int m0, int k0, int lane) {
  const float* p = base + (size_t)(m0 + (lane & 15)) * ld + k0 + ((lane >> 4) << 3);
  v4f x0 = *(const v4f*)(p);
  v4f x1 = *(const v4f*)(p + 4);
  v4f x2 = *(const v4f*)(p + 16);
  v4f x3 = *(const v4f*)(p + 20);
  v16s s;
#pragma unroll
  for (int i = 0; i < 4; ++i) {
    s[i]      = f2bf(x0[i]);
    s[i + 4]  = f2bf(x1[i]);
    s[i + 8]  = f2bf(x2[i]);
    s[i + 12] = f2bf(x3[i]);
  }
  return __builtin_bit_cast(v16bf, s);
}

#define WMMA_BF16(a, b, c) \
  __builtin_amdgcn_wmma_f32_16x16x32_bf16(false, (a), false, (b), (short)0, (c), false, false)

// ------------- K1: LayerNorm over channel dim (axis=1) -> bf16 -------------
__global__ void __launch_bounds__(256) ln_kernel(
    const float* __restrict__ q, const float* __restrict__ k, const float* __restrict__ v,
    const float* __restrict__ lnqw, const float* __restrict__ lnqb,
    const float* __restrict__ lnkw, const float* __restrict__ lnkb,
    const float* __restrict__ lnvw, const float* __restrict__ lnvb,
    unsigned short* __restrict__ qn, unsigned short* __restrict__ kn,
    unsigned short* __restrict__ vn) {
  int which = blockIdx.z;
  const float* x = (which == 0) ? q : (which == 1) ? k : v;
  const float* w = (which == 0) ? lnqw : (which == 1) ? lnkw : lnvw;
  const float* bb = (which == 0) ? lnqb : (which == 1) ? lnkb : lnvb;
  unsigned short* y = (which == 0) ? qn : (which == 1) ? kn : vn;

  int b = blockIdx.y;
  int f = blockIdx.x * 256 + threadIdx.x;
  const float* xp = x + (size_t)b * CC * FF + f;
  float s = 0.f, s2 = 0.f;
  for (int c = 0; c < CC; ++c) {
    float t = xp[(size_t)c * FF];
    s += t; s2 += t * t;
  }
  float mean = s * (1.0f / CC);
  float var  = s2 * (1.0f / CC) - mean * mean;
  float inv  = rsqrtf(var + 1e-6f);
  unsigned short* yp = y + (size_t)b * CC * FF + f;
  for (int c = 0; c < CC; ++c) {
    float t = (xp[(size_t)c * FF] - mean) * inv;
    yp[(size_t)c * FF] = f2bf(w[c] * t + bb[c]);
  }
}

// ------------- K2: weight fp32 -> bf16 -------------
__global__ void __launch_bounds__(256) wconv_kernel(
    const float* __restrict__ wq, const float* __restrict__ wk, const float* __restrict__ wv,
    unsigned short* __restrict__ oq, unsigned short* __restrict__ ok,
    unsigned short* __restrict__ ov) {
  int which = blockIdx.z;
  const float* src = (which == 0) ? wq : (which == 1) ? wk : wv;
  unsigned short* dst = (which == 0) ? oq : (which == 1) ? ok : ov;
  int i = blockIdx.x * 256 + threadIdx.x;
  dst[i] = f2bf(src[i]);
}

// ------------- K3: projection GEMM  Y = X @ W^T + bias  (bf16 out) -------------
// Block = 8 waves sharing one 64-col weight slice. Per k-step the block stages
// W[n0:n0+64, k0:k0+32] (4 KB) into LDS (async-to-LDS when available), then each
// wave computes a 32x64 tile from LDS B-frags + global A-frags.
__global__ void __launch_bounds__(256) proj_gemm(
    const unsigned short* __restrict__ qn, const unsigned short* __restrict__ kn,
    const unsigned short* __restrict__ vn,
    const unsigned short* __restrict__ wqb, const unsigned short* __restrict__ wkb,
    const unsigned short* __restrict__ wvb,
    const float* __restrict__ bq, const float* __restrict__ bk,
    const float* __restrict__ bv,
    unsigned short* __restrict__ qp, unsigned short* __restrict__ kp,
    unsigned short* __restrict__ vp) {
  int which = blockIdx.z;
  const unsigned short* X = (which == 0) ? qn : (which == 1) ? kn : vn;
  const unsigned short* W = (which == 0) ? wqb : (which == 1) ? wkb : wvb;
  const float* bias       = (which == 0) ? bq : (which == 1) ? bk : bv;
  unsigned short* Y       = (which == 0) ? qp : (which == 1) ? kp : vp;

  const int N = FF, K = FF;
  __shared__ __align__(16) unsigned short wtile[64 * 32];  // 4 KB

  int bn = blockIdx.x & 15;   // 16 n-blocks of 64
  int bm = blockIdx.x >> 4;   // 32 m-blocks of 256 rows
  int wv_ = threadIdx.x >> 5;
  int lane = threadIdx.x & 31;
  int m0 = bm * 256 + wv_ * 32;
  int n0 = bn * 64;

  // staging assignment: thread t copies 16B: W[n0+ (t>>2)][k0 + (t&3)*8 ..]
  int tid  = threadIdx.x;
  int wrow = tid >> 2;
  int wcol = (tid & 3) * 8;
  const unsigned short* wsrc = W + (size_t)(n0 + wrow) * K + wcol;
  unsigned short* wdst = &wtile[wrow * 32 + wcol];

  v8f zero = {0.f, 0.f, 0.f, 0.f, 0.f, 0.f, 0.f, 0.f};
  v8f acc[8] = {zero, zero, zero, zero, zero, zero, zero, zero};

  for (int k0 = 0; k0 < K; k0 += 32) {
#if defined(HAVE_ASYNC_LDS)
    __builtin_amdgcn_global_load_async_to_lds_b128(
        (__attribute__((address_space(1))) v4i*)(wsrc + k0),
        (__attribute__((address_space(3))) v4i*)wdst, 0, 0);
    asm volatile("s_wait_asynccnt 0x0" ::: "memory");
#else
    *(v8s*)wdst = *(const v8s*)(wsrc + k0);
#endif
    __syncthreads();

    if (k0 + 32 < K)
      __builtin_prefetch(X + (size_t)(m0 + (lane & 15)) * K + k0 + 32, 0, 1);
    v16bf a0 = load_a_bf16(X, K, m0, k0, lane);
    v16bf a1 = load_a_bf16(X, K, m0 + 16, k0, lane);
#pragma unroll
    for (int t = 0; t < 4; ++t) {
      v16bf b = load_b_bf16(wtile, 32, 16 * t, 0, lane);  // from LDS
      acc[t]     = WMMA_BF16(a0, b, acc[t]);
      acc[4 + t] = WMMA_BF16(a1, b, acc[4 + t]);
    }
    __syncthreads();
  }

  int rb = (lane >> 4) << 3;
  int c0 = lane & 15;
#pragma unroll
  for (int mi = 0; mi < 2; ++mi)
#pragma unroll
    for (int t = 0; t < 4; ++t) {
      int n = n0 + 16 * t + c0;
      float bn_ = bias[n];
#pragma unroll
      for (int i = 0; i < 8; ++i)
        Y[(size_t)(m0 + 16 * mi + rb + i) * N + n] = f2bf(acc[mi * 4 + t][i] + bn_);
    }
}

// ------------- K4: scores = (Qh/sqrt(dh)) @ Kh^T  (fp32 -> attn output) -------------
// 32x64 wave tile, register double-buffered over K (dh=128, 4 steps).
__global__ void __launch_bounds__(256) score_gemm(
    const unsigned short* __restrict__ qp, const unsigned short* __restrict__ kp,
    float* __restrict__ S) {
  int bh = blockIdx.y;
  int wave = blockIdx.x * 8 + (threadIdx.x >> 5);
  int lane = threadIdx.x & 31;
  int tn = wave & 31;   // C/64 = 32
  int tm = wave >> 5;   // C/32 = 64
  int m0 = tm * 32, n0 = tn * 64;

  const unsigned short* A  = qp + (size_t)bh * CC * DH;
  const unsigned short* Bm = kp + (size_t)bh * CC * DH;

  v8f zero = {0.f, 0.f, 0.f, 0.f, 0.f, 0.f, 0.f, 0.f};
  v8f acc[8] = {zero, zero, zero, zero, zero, zero, zero, zero};

  v16bf a0 = load_a_bf16(A, DH, m0, 0, lane);
  v16bf a1 = load_a_bf16(A, DH, m0 + 16, 0, lane);
  v16bf b0 = load_b_bf16(Bm, DH, n0, 0, lane);
  v16bf b1 = load_b_bf16(Bm, DH, n0 + 16, 0, lane);
  v16bf b2 = load_b_bf16(Bm, DH, n0 + 32, 0, lane);
  v16bf b3 = load_b_bf16(Bm, DH, n0 + 48, 0, lane);

  for (int k0 = 0; k0 < DH; k0 += 32) {
    bool more = (k0 + 32) < DH;
    v16bf na0, na1, nb0, nb1, nb2, nb3;
    if (more) {
      na0 = load_a_bf16(A, DH, m0, k0 + 32, lane);
      na1 = load_a_bf16(A, DH, m0 + 16, k0 + 32, lane);
      nb0 = load_b_bf16(Bm, DH, n0, k0 + 32, lane);
      nb1 = load_b_bf16(Bm, DH, n0 + 16, k0 + 32, lane);
      nb2 = load_b_bf16(Bm, DH, n0 + 32, k0 + 32, lane);
      nb3 = load_b_bf16(Bm, DH, n0 + 48, k0 + 32, lane);
    }
    acc[0] = WMMA_BF16(a0, b0, acc[0]);
    acc[1] = WMMA_BF16(a0, b1, acc[1]);
    acc[2] = WMMA_BF16(a0, b2, acc[2]);
    acc[3] = WMMA_BF16(a0, b3, acc[3]);
    acc[4] = WMMA_BF16(a1, b0, acc[4]);
    acc[5] = WMMA_BF16(a1, b1, acc[5]);
    acc[6] = WMMA_BF16(a1, b2, acc[6]);
    acc[7] = WMMA_BF16(a1, b3, acc[7]);
    if (more) { a0 = na0; a1 = na1; b0 = nb0; b1 = nb1; b2 = nb2; b3 = nb3; }
  }

  const float scale = 0.08838834764831845f;  // 1/sqrt(128)
  float* Sp = S + (size_t)bh * CC * CC;
  int rb = (lane >> 4) << 3;
  int c0 = lane & 15;
#pragma unroll
  for (int mi = 0; mi < 2; ++mi)
#pragma unroll
    for (int t = 0; t < 4; ++t)
#pragma unroll
      for (int i = 0; i < 8; ++i)
        Sp[(size_t)(m0 + 16 * mi + rb + i) * CC + n0 + 16 * t + c0] =
            acc[mi * 4 + t][i] * scale;
}

// ------------- K5: softmax_one in-place: e/(1 + sum e) -------------
__global__ void __launch_bounds__(256) softmax1_kernel(float* __restrict__ S) {
  __shared__ float red[256];
  int bh = blockIdx.y;
  int r  = blockIdx.x;
  float* p = S + ((size_t)bh * CC + (size_t)r) * CC;
  int tid = threadIdx.x;

  float m = -3.402823466e38f;
  for (int j = tid; j < CC; j += 256) m = fmaxf(m, p[j]);
  red[tid] = m; __syncthreads();
  for (int s = 128; s > 0; s >>= 1) {
    if (tid < s) red[tid] = fmaxf(red[tid], red[tid + s]);
    __syncthreads();
  }
  m = red[0]; __syncthreads();

  float sum = 0.f;
  for (int j = tid; j < CC; j += 256) {
    float e = expf(p[j] - m);
    p[j] = e;
    sum += e;
  }
  red[tid] = sum; __syncthreads();
  for (int s = 128; s > 0; s >>= 1) {
    if (tid < s) red[tid] += red[tid + s];
    __syncthreads();
  }
  float rinv = 1.0f / (1.0f + red[0]);
  __syncthreads();
  for (int j = tid; j < CC; j += 256) p[j] *= rinv;
}

// ------------- K6: transpose Vh (C x dh) -> Vt (dh x C) per (b,h), bf16 -------------
__global__ void __launch_bounds__(256) transpose_v(
    const unsigned short* __restrict__ vp, unsigned short* __restrict__ vt) {
  __shared__ unsigned short tile[32][33];
  int bh = blockIdx.z;
  int k0 = blockIdx.x * 32;   // over C
  int n0 = blockIdx.y * 32;   // over dh
  const unsigned short* src = vp + (size_t)bh * CC * DH;
  unsigned short* dst       = vt + (size_t)bh * DH * CC;
  int tx = threadIdx.x & 31, ty = threadIdx.x >> 5;
  for (int i = ty; i < 32; i += 8)
    tile[i][tx] = src[(size_t)(k0 + i) * DH + n0 + tx];
  __syncthreads();
  for (int i = ty; i < 32; i += 8)
    dst[(size_t)(n0 + i) * CC + k0 + tx] = tile[tx][i];
}

// ------------- K7: out = P @ Vh, (B,C,H*dh) transpose folded into stores -------------
// 32x64 wave tile, double-buffered; A-frags converted fp32->bf16 on load.
__global__ void __launch_bounds__(256) out_gemm(
    const float* __restrict__ P, const unsigned short* __restrict__ vt,
    float* __restrict__ out) {
  int bh = blockIdx.y;
  int b = bh >> 3, h = bh & 7;
  int wave = blockIdx.x * 8 + (threadIdx.x >> 5);
  int lane = threadIdx.x & 31;
  int tn = wave & 1;   // dh/64 = 2
  int tm = wave >> 1;  // C/32 = 64
  int m0 = tm * 32, n0 = tn * 64;

  const float* A           = P + (size_t)bh * CC * CC;
  const unsigned short* Bm = vt + (size_t)bh * DH * CC;

  v8f zero = {0.f, 0.f, 0.f, 0.f, 0.f, 0.f, 0.f, 0.f};
  v8f acc[8] = {zero, zero, zero, zero, zero, zero, zero, zero};

  v16bf a0 = load_a_f32(A, CC, m0, 0, lane);
  v16bf a1 = load_a_f32(A, CC, m0 + 16, 0, lane);
  v16bf b0 = load_b_bf16(Bm, CC, n0, 0, lane);
  v16bf b1 = load_b_bf16(Bm, CC, n0 + 16, 0, lane);
  v16bf b2 = load_b_bf16(Bm, CC, n0 + 32, 0, lane);
  v16bf b3 = load_b_bf16(Bm, CC, n0 + 48, 0, lane);

  for (int k0 = 0; k0 < CC; k0 += 32) {
    bool more = (k0 + 32) < CC;
    v16bf na0, na1, nb0, nb1, nb2, nb3;
    if (more) {
      __builtin_prefetch(A + (size_t)(m0 + (lane & 15)) * CC + k0 + 64, 0, 1);
      na0 = load_a_f32(A, CC, m0, k0 + 32, lane);
      na1 = load_a_f32(A, CC, m0 + 16, k0 + 32, lane);
      nb0 = load_b_bf16(Bm, CC, n0, k0 + 32, lane);
      nb1 = load_b_bf16(Bm, CC, n0 + 16, k0 + 32, lane);
      nb2 = load_b_bf16(Bm, CC, n0 + 32, k0 + 32, lane);
      nb3 = load_b_bf16(Bm, CC, n0 + 48, k0 + 32, lane);
    }
    acc[0] = WMMA_BF16(a0, b0, acc[0]);
    acc[1] = WMMA_BF16(a0, b1, acc[1]);
    acc[2] = WMMA_BF16(a0, b2, acc[2]);
    acc[3] = WMMA_BF16(a0, b3, acc[3]);
    acc[4] = WMMA_BF16(a1, b0, acc[4]);
    acc[5] = WMMA_BF16(a1, b1, acc[5]);
    acc[6] = WMMA_BF16(a1, b2, acc[6]);
    acc[7] = WMMA_BF16(a1, b3, acc[7]);
    if (more) { a0 = na0; a1 = na1; b0 = nb0; b1 = nb1; b2 = nb2; b3 = nb3; }
  }

  float* op = out + (size_t)b * CC * FF + (size_t)h * DH;
  int rb = (lane >> 4) << 3;
  int c0 = lane & 15;
#pragma unroll
  for (int mi = 0; mi < 2; ++mi)
#pragma unroll
    for (int t = 0; t < 4; ++t)
#pragma unroll
      for (int i = 0; i < 8; ++i)
        op[(size_t)(m0 + 16 * mi + rb + i) * FF + n0 + 16 * t + c0] =
            acc[mi * 4 + t][i];
}

// ---------------- workspace layout (ushort elements) ----------------
static constexpr size_t EL = (size_t)BB * CC * FF;     // 8,388,608 elems
static constexpr size_t QN_OFF = 0;
static constexpr size_t KN_OFF = QN_OFF + EL;
static constexpr size_t VN_OFF = KN_OFF + EL;
static constexpr size_t WQ_OFF = VN_OFF + EL;
static constexpr size_t WK_OFF = WQ_OFF + (size_t)FF * FF;
static constexpr size_t WV_OFF = WK_OFF + (size_t)FF * FF;
static constexpr size_t QP_OFF = WV_OFF + (size_t)FF * FF;
static constexpr size_t KP_OFF = QP_OFF + EL;
static constexpr size_t VP_OFF = KP_OFF + EL;
static constexpr size_t VT_OFF = VP_OFF + EL;
// total ~ 123.8 MB of bf16 scratch

extern "C" void kernel_launch(void* const* d_in, const int* in_sizes, int n_in,
                              void* d_out, int out_size, void* d_ws, size_t ws_size,
                              hipStream_t stream) {
  const float* q    = (const float*)d_in[0];
  const float* k    = (const float*)d_in[1];
  const float* v    = (const float*)d_in[2];
  const float* lnqw = (const float*)d_in[3];
  const float* lnqb = (const float*)d_in[4];
  const float* lnkw = (const float*)d_in[5];
  const float* lnkb = (const float*)d_in[6];
  const float* lnvw = (const float*)d_in[7];
  const float* lnvb = (const float*)d_in[8];
  const float* wq   = (const float*)d_in[9];
  const float* bq   = (const float*)d_in[10];
  const float* wk   = (const float*)d_in[11];
  const float* bk   = (const float*)d_in[12];
  const float* wv   = (const float*)d_in[13];
  const float* bv   = (const float*)d_in[14];

  float* out  = (float*)d_out;           // (B,C,F)
  float* attn = out + EL;                // (B,H,C,C)
  unsigned short* ws = (unsigned short*)d_ws;

  // 1) layernorm q/k/v -> bf16
  ln_kernel<<<dim3(FF / 256, BB, 3), 256, 0, stream>>>(
      q, k, v, lnqw, lnqb, lnkw, lnkb, lnvw, lnvb,
      ws + QN_OFF, ws + KN_OFF, ws + VN_OFF);

  // 2) weights -> bf16
  wconv_kernel<<<dim3((FF * FF) / 256, 1, 3), 256, 0, stream>>>(
      wq, wk, wv, ws + WQ_OFF, ws + WK_OFF, ws + WV_OFF);

  // 3) projections (WMMA, LDS-staged weights, async-to-LDS when available)
  proj_gemm<<<dim3(512, 1, 3), 256, 0, stream>>>(
      ws + QN_OFF, ws + KN_OFF, ws + VN_OFF,
      ws + WQ_OFF, ws + WK_OFF, ws + WV_OFF,
      bq, bk, bv,
      ws + QP_OFF, ws + KP_OFF, ws + VP_OFF);

  // 4) scores -> attn output region (WMMA, double-buffered 32x64 tiles)
  score_gemm<<<dim3(256, BB * HH), 256, 0, stream>>>(
      ws + QP_OFF, ws + KP_OFF, attn);

  // 5) softmax_one in-place on attn
  softmax1_kernel<<<dim3(CC, BB * HH), 256, 0, stream>>>(attn);

  // 6) transpose V per head (independent; bf16)
  transpose_v<<<dim3(CC / 32, DH / 32, BB * HH), 256, 0, stream>>>(
      ws + VP_OFF, ws + VT_OFF);

  // 7) out = attn @ V (WMMA, double-buffered 32x64 tiles)
  out_gemm<<<dim3(16, BB * HH), 256, 0, stream>>>(
      attn, ws + VT_OFF, out);
}
